// Stage_18906446037285
// MI455X (gfx1250) — matrix operand order repaired and checked by reading, not compiled, not run on hardware
//
#include <hip/hip_runtime.h>
#include <hip/hip_bf16.h>
#include <cstdint>

// Problem constants (match reference)
#define BB   4
#define NN   16384
#define CCH  256
#define KNB  16
#define DDEP 4
#define HHID 512
#define MROWS (BB*NN)      // 65536 rows
#define EPSV 1e-5f

typedef __attribute__((ext_vector_type(16))) __bf16 v16bf;
typedef __attribute__((ext_vector_type(8)))  float  v8f;
typedef __attribute__((address_space(3)))    __bf16 lds_bf16;

// ---------------------------------------------------------------------------
// Branchless fast GELU (tanh form): TRANS v_exp_f32 + v_rcp_f32, no exec-mask
// divergence.
// ---------------------------------------------------------------------------
__device__ __forceinline__ float gelu_fast(float v) {
  float y = 0.7978845608028654f * v * (1.0f + 0.044715f * v * v);
  float e = __expf(2.0f * y);                               // v_exp_f32
  float t = 1.0f - 2.0f * __builtin_amdgcn_rcpf(e + 1.0f);  // tanh(y)
  return 0.5f * v * (1.0f + t);
}

// ---------------------------------------------------------------------------
// WMMA A-fragment loaders (wave32 layout per CDNA5 ISA 7.12.2):
// A 16x32 bf16: element e of v16bf: e in [0,8) -> K=h*8+e, e in [8,16) ->
// K=16+h*8+(e-8); lanes 0-15 rows M=lane (K lo half), lanes 16-31 same rows.
// ---------------------------------------------------------------------------
__device__ __forceinline__ v16bf load_a_f32(const float* X, int ld,
                                            int m0, int k0, int lane) {
  const int r = lane & 15, h = (lane >> 4) & 1;
  const float* p = X + (size_t)(m0 + r) * ld + k0 + h * 8;
  v16bf a;
#pragma unroll
  for (int j = 0; j < 4; ++j) {
    float2 lo = *(const float2*)(p + 2 * j);
    float2 hi = *(const float2*)(p + 16 + 2 * j);
    a[2 * j]     = (__bf16)lo.x;  a[2 * j + 1] = (__bf16)lo.y;
    a[8 + 2 * j] = (__bf16)hi.x;  a[9 + 2 * j] = (__bf16)hi.y;
  }
  return a;
}

__device__ __forceinline__ v16bf load_a_bf16(const __bf16* X, int ld,
                                             int m0, int k0, int lane) {
  const int r = lane & 15, h = (lane >> 4) & 1;
  const __bf16* p = X + (size_t)(m0 + r) * ld + k0 + h * 8;
  v16bf a;
#pragma unroll
  for (int j = 0; j < 8; ++j) {
    a[j]     = p[j];        // 16B contiguous
    a[8 + j] = p[16 + j];   // 16B contiguous
  }
  return a;
}

// ---------------------------------------------------------------------------
// WMMA GEMM: Out[M x Ncols] = A[M x KD] * W[KD x Ncols]
// Weights pre-transposed+converted: Wt is bf16 [Ncols, KD].
// Block tile 128(M) x 128(N); 8 waves as 4(M) x 2(N); each wave owns a
// 32(M) x 64(N) tile = 2 M-tiles x 4 N-tiles -> 8 WMMAs per K-step with only
// 2 A fragments (4 WMMAs per global A fragment, B reuse via LDS).
// The block's whole B panel (128 cols x KD bf16) is staged into LDS once via
// global_load_async_to_lds_b128 (ASYNCcnt + s_wait_asynccnt), then B
// fragments are two contiguous conflict-free ds_load_b128 each (row pitch
// KD+8 elements -> lane c hits banks 4c..4c+3).
// MODE 0: out_bf16 = bf16(gelu(acc + bias[col]));  MODE 1: out_f32 = acc
// ---------------------------------------------------------------------------
template <int KD, bool A_BF16, int MODE>
__global__ __launch_bounds__(256)
void gemm_wmma_bf16(const void* __restrict__ Aptr,
                    const __bf16* __restrict__ Wt,
                    const float* __restrict__ bias,
                    void* __restrict__ Out,
                    int Ncols) {
  constexpr int KPAD = KD + 8;
  __shared__ __bf16 Bs[128 * KPAD];

  const int lane = threadIdx.x & 31;
  const int wave = threadIdx.x >> 5;
  const int wm = wave & 3, wn = wave >> 2;
  const int m0 = blockIdx.y * 128 + wm * 32;    // wave's first row (32 rows)
  const int n0 = blockIdx.x * 128;              // block's first output column
  const int nw = n0 + wn * 64;                  // wave's first column (64 cols)

  // ---- async-stage entire B panel: cols n0..n0+127, all KD, bf16 ----
  {
    constexpr int CPC = KD / 8;                 // 16B chunks per column
    constexpr int TOTAL = 128 * CPC;
#pragma unroll
    for (int id = threadIdx.x; id < TOTAL; id += 256) {
      const int col  = id / CPC;
      const int koff = (id - col * CPC) * 8;
      const unsigned gofs = (unsigned)(((unsigned)(n0 + col) * KD + koff) * 2);
      const unsigned lofs =
          (unsigned)(uintptr_t)(lds_bf16*)&Bs[col * KPAD + koff];
      asm volatile("global_load_async_to_lds_b128 %0, %1, %2"
                   :: "v"(lofs), "v"(gofs), "s"(Wt) : "memory");
    }
    asm volatile("s_wait_asynccnt 0x0" ::: "memory");
    __syncthreads();
  }

  v8f acc[2][4];
#pragma unroll
  for (int tm = 0; tm < 2; ++tm)
#pragma unroll
    for (int tn = 0; tn < 4; ++tn)
      acc[tm][tn] = v8f{};

  const int h = (lane >> 4) & 1, cl = lane & 15;

  for (int k0 = 0; k0 < KD; k0 += 32) {
    // unconditional speculative prefetch of next A K-slab (no branch; a
    // translation fault on the tail iteration is silently dropped)
    {
      const char* ap = (const char*)Aptr;
      const size_t esz = A_BF16 ? 2 : 4;
      __builtin_prefetch(ap + ((size_t)(m0 + (lane & 15)) * KD + k0 + 32) * esz,
                         0, 3);
    }

    v16bf a0, a1;
    if (A_BF16) {
      a0 = load_a_bf16((const __bf16*)Aptr, KD, m0, k0, lane);
      a1 = load_a_bf16((const __bf16*)Aptr, KD, m0 + 16, k0, lane);
    } else {
      a0 = load_a_f32((const float*)Aptr, KD, m0, k0, lane);
      a1 = load_a_f32((const float*)Aptr, KD, m0 + 16, k0, lane);
    }

    // B 32x16: element j of v16bf <-> K = h*8 + j (j<8), 16 + h*8 + (j-8) (j>=8), col = cl
    v16bf bfr[4];
#pragma unroll
    for (int tn = 0; tn < 4; ++tn) {
#pragma unroll
      for (int j = 0; j < 16; ++j)
        bfr[tn][j] = Bs[(wn * 64 + tn * 16 + cl) * KPAD + k0 + h * 8 + (j < 8 ? j : j + 8)];
    }

#pragma unroll
    for (int tn = 0; tn < 4; ++tn) {
      acc[0][tn] = __builtin_amdgcn_wmma_f32_16x16x32_bf16(
          false, a0, false, bfr[tn], (short)0, acc[0][tn], false, false);
      acc[1][tn] = __builtin_amdgcn_wmma_f32_16x16x32_bf16(
          false, a1, false, bfr[tn], (short)0, acc[1][tn], false, false);
    }
  }

  // C/D 16x16 f32: lanes 0-15: VGPR i <-> M=i ; lanes 16-31: M=8+i ; N=lane&15
#pragma unroll
  for (int tm = 0; tm < 2; ++tm) {
#pragma unroll
    for (int tn = 0; tn < 4; ++tn) {
#pragma unroll
      for (int i = 0; i < 8; ++i) {
        const int row = m0 + tm * 16 + h * 8 + i;
        const int col = nw + tn * 16 + cl;
        float v = acc[tm][tn][i];
        if (MODE == 0) {
          v = gelu_fast(v + bias[col]);
          ((__bf16*)Out)[(size_t)row * Ncols + col] = (__bf16)v;
        } else {
          ((float*)Out)[(size_t)row * Ncols + col] = v;
        }
      }
    }
  }
}

// ---------------------------------------------------------------------------
// Weight prep: f32 [K,N] row-major -> bf16 [N,K] (so B fragments are two
// contiguous 16B reads). Weights are tiny (<= 512KB each), runs once/launch.
// ---------------------------------------------------------------------------
__global__ __launch_bounds__(256)
void transpose_cvt_bf16(const float* __restrict__ W, __bf16* __restrict__ Wt,
                        int K, int N) {
  const int t = blockIdx.x * 256 + threadIdx.x;
  if (t >= K * N) return;
  const int k = t / N, n = t - k * N;
  Wt[(size_t)n * K + k] = (__bf16)W[t];
}

// ---------------------------------------------------------------------------
// LFP neighbor gather: y[b,n,c] = max_k p[b, knn[b,n,k], c] - p[b,n,c]
// 64 threads share a point -> each neighbor row is one coalesced 1KB read
// (p is 64MB, L2-resident on the 192MB L2).
// ---------------------------------------------------------------------------
__global__ __launch_bounds__(256)
void lfp_gather_max(const float* __restrict__ p,
                    const int* __restrict__ knn,
                    float* __restrict__ y) {
  const int t  = blockIdx.x * 256 + threadIdx.x;   // B*N*64 threads
  const int cg = t & 63;
  const int bn = t >> 6;
  const int b  = bn >> 14;                          // N = 16384
  const int* kn = knn + (size_t)bn * KNB;
  const float4* pb = (const float4*)p + (size_t)b * NN * 64;

  float4 mx = make_float4(-3.402823466e38f, -3.402823466e38f,
                          -3.402823466e38f, -3.402823466e38f);
#pragma unroll
  for (int k = 0; k < KNB; ++k) {
    const int idx = kn[k];
    float4 v = pb[(size_t)idx * 64 + cg];
    mx.x = fmaxf(mx.x, v.x); mx.y = fmaxf(mx.y, v.y);
    mx.z = fmaxf(mx.z, v.z); mx.w = fmaxf(mx.w, v.w);
  }
  float4 s = ((const float4*)p)[(size_t)bn * 64 + cg];
  ((float4*)y)[(size_t)bn * 64 + cg] =
      make_float4(mx.x - s.x, mx.y - s.y, mx.z - s.z, mx.w - s.w);
}

// ---------------------------------------------------------------------------
// Deterministic BatchNorm stats (two-pass partial reduction, biased var)
// ---------------------------------------------------------------------------
__global__ __launch_bounds__(256)
void colstats_partial(const float* __restrict__ y, float* __restrict__ partial) {
  const int c = threadIdx.x;          // 256 channels
  const int blk = blockIdx.x;         // 256 blocks x 256 rows
  float s = 0.f, ss = 0.f;
  for (int r = 0; r < 256; ++r) {
    float v = y[((size_t)blk * 256 + r) * CCH + c];
    s += v; ss += v * v;
  }
  partial[(size_t)blk * 512 + c]       = s;
  partial[(size_t)blk * 512 + 256 + c] = ss;
}

__global__ __launch_bounds__(256)
void colstats_finalize(const float* __restrict__ partial,
                       const float* __restrict__ bnw,
                       const float* __restrict__ bnb,
                       float* __restrict__ scale,
                       float* __restrict__ shift) {
  const int c = threadIdx.x;
  float s = 0.f, ss = 0.f;
  for (int blk = 0; blk < 256; ++blk) {
    s  += partial[(size_t)blk * 512 + c];
    ss += partial[(size_t)blk * 512 + 256 + c];
  }
  const float inv = 1.0f / (float)MROWS;
  const float m = s * inv;
  const float v = ss * inv - m * m;      // biased variance, matches reference
  const float sc = rsqrtf(v + EPSV) * bnw[c];
  scale[c] = sc;
  shift[c] = bnb[c] - m * sc;
}

__global__ __launch_bounds__(256)
void bn_apply_residual(float* __restrict__ x, const float* __restrict__ y,
                       const float* __restrict__ scale,
                       const float* __restrict__ shift) {
  const int t = blockIdx.x * 256 + threadIdx.x;    // MROWS*64 threads
  const int cg = t & 63;
  float4 sc = ((const float4*)scale)[cg];
  float4 sh = ((const float4*)shift)[cg];
  float4 yv = ((const float4*)y)[t];
  float4 xv = ((float4*)x)[t];
  xv.x += yv.x * sc.x + sh.x;
  xv.y += yv.y * sc.y + sh.y;
  xv.z += yv.z * sc.z + sh.z;
  xv.w += yv.w * sc.w + sh.w;
  ((float4*)x)[t] = xv;
}

__global__ __launch_bounds__(256)
void copy_f32x4(const float4* __restrict__ src, float4* __restrict__ dst) {
  const int t = blockIdx.x * 256 + threadIdx.x;
  dst[t] = src[t];
}

// ---------------------------------------------------------------------------
// Host-side stage helpers
// ---------------------------------------------------------------------------
namespace {

struct Ws {
  __bf16* hbuf;    // [MROWS, HHID] bf16   (overlaid with pbuf)
  float*  pbuf;    // [MROWS, CCH]  f32    (overlaid with hbuf)
  float*  ybuf;    // [MROWS, CCH]  f32
  float*  partial; // [256, 512]
  float*  scale;   // [256]
  float*  shift;   // [256]
  __bf16* lfp_wt;  // [D][CCH, CCH]  (bf16, [N,K])
  __bf16* w1t;     // [3][HHID, CCH] (mlp0 then mlps j=0,1)
  __bf16* w2t;     // [3][CCH, HHID]
};

inline void run_stats(const float* y, const float* bnw, const float* bnb,
                      const Ws& w, hipStream_t s) {
  colstats_partial<<<256, 256, 0, s>>>(y, w.partial);
  colstats_finalize<<<1, 256, 0, s>>>(w.partial, bnw, bnb, w.scale, w.shift);
}

inline void run_mlp(float* x, const __bf16* w1t, const float* b1,
                    const __bf16* w2t, const float* bnw, const float* bnb,
                    const Ws& w, hipStream_t s) {
  dim3 g1(HHID / 128, MROWS / 128);   // 4 x 512
  gemm_wmma_bf16<CCH, false, 0><<<g1, 256, 0, s>>>(x, w1t, b1, w.hbuf, HHID);
  dim3 g2(CCH / 128, MROWS / 128);    // 2 x 512
  gemm_wmma_bf16<HHID, true, 1><<<g2, 256, 0, s>>>(w.hbuf, w2t, nullptr,
                                                   w.ybuf, CCH);
  run_stats(w.ybuf, bnw, bnb, w, s);
  bn_apply_residual<<<MROWS * 64 / 256, 256, 0, s>>>(x, w.ybuf, w.scale, w.shift);
}

inline void run_lfp(float* x, const int* knn, const __bf16* lwt,
                    const float* bnw, const float* bnb, const Ws& w,
                    hipStream_t s) {
  dim3 g(CCH / 128, MROWS / 128);
  gemm_wmma_bf16<CCH, false, 1><<<g, 256, 0, s>>>(x, lwt, nullptr, w.pbuf, CCH);
  lfp_gather_max<<<MROWS * 64 / 256, 256, 0, s>>>(w.pbuf, knn, w.ybuf);
  run_stats(w.ybuf, bnw, bnb, w, s);
  bn_apply_residual<<<MROWS * 64 / 256, 256, 0, s>>>(x, w.ybuf, w.scale, w.shift);
}

}  // namespace

extern "C" void kernel_launch(void* const* d_in, const int* in_sizes, int n_in,
                              void* d_out, int out_size, void* d_ws, size_t ws_size,
                              hipStream_t stream) {
  const float* x        = (const float*)d_in[0];
  const int*   knn      = (const int*)d_in[1];
  const float* lfp_w    = (const float*)d_in[2];   // [D,C,C]
  const float* lfp_bnw  = (const float*)d_in[3];   // [D,C]
  const float* lfp_bnb  = (const float*)d_in[4];   // [D,C]
  const float* mlp0_w1  = (const float*)d_in[5];   // [C,H]
  const float* mlp0_b1  = (const float*)d_in[6];   // [H]
  const float* mlp0_w2  = (const float*)d_in[7];   // [H,C]
  const float* mlp0_bnw = (const float*)d_in[8];
  const float* mlp0_bnb = (const float*)d_in[9];
  const float* mlps_w1  = (const float*)d_in[10];  // [D/2,C,H]
  const float* mlps_b1  = (const float*)d_in[11];  // [D/2,H]
  const float* mlps_w2  = (const float*)d_in[12];  // [D/2,H,C]
  const float* mlps_bnw = (const float*)d_in[13];  // [D/2,C]
  const float* mlps_bnb = (const float*)d_in[14];  // [D/2,C]

  float* xout = (float*)d_out;                     // running x lives in d_out

  // Workspace carve-up (hbuf/pbuf overlaid: never live simultaneously)
  char* ws = (char*)d_ws;
  Ws w;
  w.hbuf    = (__bf16*)ws;                                   // 64 MB
  w.pbuf    = (float*)ws;                                    // 64 MB (overlay)
  w.ybuf    = (float*)(ws + (size_t)67108864);               // 64 MB
  w.partial = (float*)(ws + (size_t)134217728);              // 512 KB
  w.scale   = (float*)(ws + (size_t)134744064);
  w.shift   = (float*)(ws + (size_t)134745088);
  char* wt  = ws + (size_t)134746112;                        // bf16 weights ~2MB
  w.lfp_wt  = (__bf16*)wt;                                   // 4*65536
  w.w1t     = (__bf16*)(wt + (size_t)4 * CCH * CCH * 2);     // 3*131072
  w.w2t     = w.w1t + (size_t)3 * CCH * HHID;                // 3*131072

  // ---- one-time (per launch) weight convert+transpose to bf16 [N,K] ----
  for (int i = 0; i < DDEP; ++i)
    transpose_cvt_bf16<<<(CCH * CCH + 255) / 256, 256, 0, stream>>>(
        lfp_w + (size_t)i * CCH * CCH, w.lfp_wt + (size_t)i * CCH * CCH,
        CCH, CCH);
  transpose_cvt_bf16<<<(CCH * HHID + 255) / 256, 256, 0, stream>>>(
      mlp0_w1, w.w1t, CCH, HHID);
  transpose_cvt_bf16<<<(CCH * HHID + 255) / 256, 256, 0, stream>>>(
      mlp0_w2, w.w2t, HHID, CCH);
  for (int j = 0; j < 2; ++j) {
    transpose_cvt_bf16<<<(CCH * HHID + 255) / 256, 256, 0, stream>>>(
        mlps_w1 + (size_t)j * CCH * HHID,
        w.w1t + (size_t)(j + 1) * CCH * HHID, CCH, HHID);
    transpose_cvt_bf16<<<(CCH * HHID + 255) / 256, 256, 0, stream>>>(
        mlps_w2 + (size_t)j * HHID * CCH,
        w.w2t + (size_t)(j + 1) * CCH * HHID, HHID, CCH);
  }

  // x -> out (never mutate inputs)
  copy_f32x4<<<MROWS * 64 / 256, 256, 0, stream>>>((const float4*)x,
                                                   (float4*)xout);

  // x += mlp0(x)
  run_mlp(xout, w.w1t, mlp0_b1, w.w2t, mlp0_bnw, mlp0_bnb, w, stream);

  for (int i = 0; i < DDEP; ++i) {
    run_lfp(xout, knn, w.lfp_wt + (size_t)i * CCH * CCH,
            lfp_bnw + (size_t)i * CCH, lfp_bnb + (size_t)i * CCH, w, stream);
    if (i & 1) {
      const int j = i >> 1;
      run_mlp(xout, w.w1t + (size_t)(j + 1) * CCH * HHID,
              mlps_b1 + (size_t)j * HHID,
              w.w2t + (size_t)(j + 1) * CCH * HHID,
              mlps_bnw + (size_t)j * CCH, mlps_bnb + (size_t)j * CCH,
              w, stream);
    }
  }
}
